// SAModule_24953759990274
// MI455X (gfx1250) — matrix operand order repaired and checked by reading, not compiled, not run on hardware
//
#include <hip/hip_runtime.h>
#include <hip/hip_bf16.h>

#define NB    8
#define NPC   8192
#define SS    2048
#define NC    (NB * SS)          // 16384 centers
#define KNB   32
#define CIN   64
#define DPAD  96                 // 67 padded to 96 for K-steps of 32
#define RAD2  0.04f
#define FINF  3.402823e38f

typedef __attribute__((ext_vector_type(16))) __bf16 v16bf;
typedef __attribute__((ext_vector_type(8)))  __bf16 v8bf;
typedef __attribute__((ext_vector_type(8)))  float  v8f;

__device__ __forceinline__ unsigned short f2bf(float f) {
    unsigned u = __float_as_uint(f);
    u += 0x7FFFu + ((u >> 16) & 1u);      // round-to-nearest-even
    return (unsigned short)(u >> 16);
}

// A fragment (16x32 bf16): lane<16 holds K {0..7,16..23}, lane>=16 holds K {8..15,24..31}
__device__ __forceinline__ v16bf ldsA(const unsigned short* base, int rs, int mt, int ks, int lane) {
    int row = mt * 16 + (lane & 15);
    int hi8 = (lane >= 16) ? 8 : 0;
    const unsigned short* p = base + row * rs + ks * 32 + hi8;
    v8bf lo = *(const v8bf*)p;
    v8bf hi = *(const v8bf*)(p + 16);
    return __builtin_shufflevector(lo, hi, 0,1,2,3,4,5,6,7,8,9,10,11,12,13,14,15);
}

// B fragment (32x16 bf16) from [N][K]-transposed weights: lane<16 holds K {0..7,16..23}, lane>=16 holds K {8..15,24..31}
__device__ __forceinline__ v16bf ldsB(const unsigned short* wt, int kd, int nt, int ks, int lane) {
    int n  = nt * 16 + (lane & 15);
    int kb = ks * 32 + ((lane >= 16) ? 8 : 0);
    const unsigned short* p = wt + n * kd + kb;
    v8bf lo = *(const v8bf*)p;
    v8bf hi = *(const v8bf*)(p + 16);
    return __builtin_shufflevector(lo, hi, 0,1,2,3,4,5,6,7,8,9,10,11,12,13,14,15);
}

// ---------------- Kernel 1: farthest point sampling (1 block per cloud) -------------
__global__ __launch_bounds__(1024) void fps_kernel(const float* __restrict__ pos,
                                                   float* __restrict__ outPos,
                                                   float* __restrict__ outBatch) {
    const int b   = blockIdx.x;
    const int tid = threadIdx.x;
    const float* pb = pos + (size_t)b * NPC * 3;

    float px[8], py[8], pz[8], mind[8];
#pragma unroll
    for (int k = 0; k < 8; ++k) {
        int i = tid + k * 1024;
        px[k] = pb[i * 3 + 0]; py[k] = pb[i * 3 + 1]; pz[k] = pb[i * 3 + 2];
        mind[k] = FINF;
    }
    __shared__ float qc[3];
    __shared__ float wv[32];
    __shared__ int   wi[32];

    for (int i = tid; i < SS; i += 1024) outBatch[b * SS + i] = (float)b;
    if (tid == 0) {
        qc[0] = pb[0]; qc[1] = pb[1]; qc[2] = pb[2];
        size_t o = (size_t)(b * SS) * 3;
        outPos[o + 0] = pb[0]; outPos[o + 1] = pb[1]; outPos[o + 2] = pb[2];
    }
    __syncthreads();

    for (int t = 1; t < SS; ++t) {
        float qx = qc[0], qy = qc[1], qz = qc[2];
        float bv = -1.f; int bid = 0;
#pragma unroll
        for (int k = 0; k < 8; ++k) {
            float dx = px[k] - qx, dy = py[k] - qy, dz = pz[k] - qz;
            float d = dx * dx + dy * dy + dz * dz;
            mind[k] = fminf(mind[k], d);
            if (mind[k] > bv) { bv = mind[k]; bid = tid + k * 1024; }
        }
#pragma unroll
        for (int off = 16; off > 0; off >>= 1) {        // wave32 argmax
            float ov = __shfl_xor(bv, off);
            int   oi = __shfl_xor(bid, off);
            if (ov > bv || (ov == bv && oi < bid)) { bv = ov; bid = oi; }
        }
        if ((tid & 31) == 0) { wv[tid >> 5] = bv; wi[tid >> 5] = bid; }
        __syncthreads();
        if (tid == 0) {
            float gv = wv[0]; int gi = wi[0];
            for (int j = 1; j < 32; ++j)
                if (wv[j] > gv || (wv[j] == gv && wi[j] < gi)) { gv = wv[j]; gi = wi[j]; }
            float nx = pb[gi * 3 + 0], ny = pb[gi * 3 + 1], nz = pb[gi * 3 + 2];
            qc[0] = nx; qc[1] = ny; qc[2] = nz;
            size_t o = (size_t)(b * SS + t) * 3;
            outPos[o + 0] = nx; outPos[o + 1] = ny; outPos[o + 2] = nz;
        }
        __syncthreads();
    }
}

// ---------------- Kernel 2: K nearest within radius (1 wave per center) -------------
__global__ __launch_bounds__(256) void radius_kernel(const float* __restrict__ pos,
                                                     const float* __restrict__ qpos,
                                                     int* __restrict__ nidx) {
    const int wid  = (blockIdx.x * 256 + threadIdx.x) >> 5;
    const int lane = threadIdx.x & 31;
    if (wid >= NC) return;
    const int b = wid / SS;
    const float* pb = pos + (size_t)b * NPC * 3;
    const float qx = qpos[wid * 3 + 0], qy = qpos[wid * 3 + 1], qz = qpos[wid * 3 + 2];

    float bd[KNB]; int bi[KNB];
#pragma unroll
    for (int j = 0; j < KNB; ++j) { bd[j] = FINF; bi[j] = 0x7FFFFFFF; }
    float cmax = FINF; int cpos = 0;

    for (int t = 0; t < NPC / 32; ++t) {                 // each lane scans 256 points
        int pid = t * 32 + lane;
        float dx = pb[pid * 3 + 0] - qx;
        float dy = pb[pid * 3 + 1] - qy;
        float dz = pb[pid * 3 + 2] - qz;
        float d2 = dx * dx + dy * dy + dz * dz;
        if (d2 <= RAD2 && d2 < cmax) {
#pragma unroll
            for (int j = 0; j < KNB; ++j) if (j == cpos) { bd[j] = d2; bi[j] = pid; }
            cmax = -1.f;
#pragma unroll
            for (int j = 0; j < KNB; ++j) if (bd[j] > cmax) { cmax = bd[j]; cpos = j; }
        }
    }
    // 32 rounds of cross-lane min-extraction -> globally sorted K nearest
    for (int r = 0; r < KNB; ++r) {
        float lmd = FINF; int lmi = 0x7FFFFFFF; int lmp = 0;
#pragma unroll
        for (int j = 0; j < KNB; ++j) {
            bool better = (bd[j] < lmd) || (bd[j] == lmd && bi[j] < lmi);
            if (better) { lmd = bd[j]; lmi = bi[j]; lmp = j; }
        }
        float gd = lmd; int gi = lmi;
#pragma unroll
        for (int off = 16; off > 0; off >>= 1) {
            float od = __shfl_xor(gd, off);
            int   oi = __shfl_xor(gi, off);
            if (od < gd || (od == gd && oi < gi)) { gd = od; gi = oi; }
        }
        if (gd == lmd && gi == lmi) {                    // owner removes its entry
#pragma unroll
            for (int j = 0; j < KNB; ++j) if (j == lmp) { bd[j] = FINF; bi[j] = 0x7FFFFFFF; }
        }
        if (lane == r) nidx[wid * KNB + r] = (gd < 1.0e38f) ? (b * NPC + gi) : -1;
    }
}

// ---------------- Kernel 3: gather + bf16-WMMA MLP + masked max (1 wave/center) -----
__global__ __launch_bounds__(256) void pointconv_kernel(
        const float* __restrict__ x,   const float* __restrict__ pos,
        const float* __restrict__ qpos, const int* __restrict__ nidx,
        const float* __restrict__ W1, const float* __restrict__ b1,
        const float* __restrict__ W2, const float* __restrict__ b2,
        const float* __restrict__ W3, const float* __restrict__ b3,
        float* __restrict__ outX) {
    __shared__ unsigned short sW1[64 * DPAD];     // W1^T [n][k], k padded 67->96
    __shared__ unsigned short sW2[64 * 64];       // W2^T [n][k]
    __shared__ unsigned short sW3[128 * 64];      // W3^T [n][k]
    __shared__ unsigned short sA[8][32 * DPAD];   // per-wave activations (layer input)
    __shared__ unsigned short sH[8][32 * 64];     // per-wave hidden

    const int tid = threadIdx.x;
    for (int i = tid; i < 64 * DPAD; i += 256) {
        int n = i / DPAD, k = i % DPAD;
        sW1[n * DPAD + k] = f2bf((k < 67) ? W1[k * 64 + n] : 0.f);
    }
    for (int i = tid; i < 64 * 64; i += 256) {
        int n = i / 64, k = i % 64;
        sW2[n * 64 + k] = f2bf(W2[k * 64 + n]);
    }
    for (int i = tid; i < 128 * 64; i += 256) {
        int n = i / 64, k = i % 64;
        sW3[n * 64 + k] = f2bf(W3[k * 128 + n]);
    }
    __syncthreads();

    const int lane = tid & 31;
    const int lw   = tid >> 5;
    const int gw   = blockIdx.x * 8 + lw;          // 0..2047 global wave id
    unsigned short* bufA = sA[lw];
    unsigned short* bufH = sH[lw];

    for (int c = gw; c < NC; c += 2048) {
        // ---- gather: lane j loads neighbor j (64 feats + relpos) into bufA row j ----
        int gid = nidx[c * KNB + lane];
        bool valid = gid >= 0;
        unsigned vmask = (unsigned)__ballot(valid);
        int cloudBase = (c / SS) * NPC;
        int g = valid ? gid : cloudBase;
        float qx = qpos[c * 3 + 0], qy = qpos[c * 3 + 1], qz = qpos[c * 3 + 2];
        const float4* xr = (const float4*)(x + (size_t)g * CIN);
#pragma unroll
        for (int t = 0; t < 8; ++t) {
            float4 a = xr[2 * t + 0];
            float4 e = xr[2 * t + 1];
            uint4 pk;
            pk.x = (unsigned)f2bf(a.x) | ((unsigned)f2bf(a.y) << 16);
            pk.y = (unsigned)f2bf(a.z) | ((unsigned)f2bf(a.w) << 16);
            pk.z = (unsigned)f2bf(e.x) | ((unsigned)f2bf(e.y) << 16);
            pk.w = (unsigned)f2bf(e.z) | ((unsigned)f2bf(e.w) << 16);
            *(uint4*)&bufA[lane * DPAD + t * 8] = pk;
        }
        uint4 zz = make_uint4(0, 0, 0, 0);
        *(uint4*)&bufA[lane * DPAD + 64] = zz;
        *(uint4*)&bufA[lane * DPAD + 72] = zz;
        *(uint4*)&bufA[lane * DPAD + 80] = zz;
        *(uint4*)&bufA[lane * DPAD + 88] = zz;
        float pjx = pos[(size_t)g * 3 + 0], pjy = pos[(size_t)g * 3 + 1], pjz = pos[(size_t)g * 3 + 2];
        bufA[lane * DPAD + 64] = f2bf(pjx - qx);
        bufA[lane * DPAD + 65] = f2bf(pjy - qy);
        bufA[lane * DPAD + 66] = f2bf(pjz - qz);
        asm volatile("s_wait_dscnt 0x0" ::: "memory");

        // ---- layer 1: [32x96] @ [96x64], ReLU -> bufH ----
        for (int nt = 0; nt < 4; ++nt) {
            float bias = b1[nt * 16 + (lane & 15)];
            for (int mt = 0; mt < 2; ++mt) {
                v8f acc = {};
#pragma unroll
                for (int ks = 0; ks < 3; ++ks) {
                    v16bf af = ldsA(bufA, DPAD, mt, ks, lane);
                    v16bf bf = ldsB(sW1, DPAD, nt, ks, lane);
                    acc = __builtin_amdgcn_wmma_f32_16x16x32_bf16(false, af, false, bf,
                                                                  (short)0, acc, false, false);
                }
                int m0 = mt * 16 + ((lane >= 16) ? 8 : 0);
                int n  = nt * 16 + (lane & 15);
#pragma unroll
                for (int r = 0; r < 8; ++r) {
                    float v = acc[r] + bias;
                    v = v > 0.f ? v : 0.f;
                    bufH[(m0 + r) * 64 + n] = f2bf(v);
                }
            }
        }
        asm volatile("s_wait_dscnt 0x0" ::: "memory");

        // ---- layer 2: [32x64] @ [64x64], ReLU -> bufA (reused) ----
        for (int nt = 0; nt < 4; ++nt) {
            float bias = b2[nt * 16 + (lane & 15)];
            for (int mt = 0; mt < 2; ++mt) {
                v8f acc = {};
#pragma unroll
                for (int ks = 0; ks < 2; ++ks) {
                    v16bf af = ldsA(bufH, 64, mt, ks, lane);
                    v16bf bf = ldsB(sW2, 64, nt, ks, lane);
                    acc = __builtin_amdgcn_wmma_f32_16x16x32_bf16(false, af, false, bf,
                                                                  (short)0, acc, false, false);
                }
                int m0 = mt * 16 + ((lane >= 16) ? 8 : 0);
                int n  = nt * 16 + (lane & 15);
#pragma unroll
                for (int r = 0; r < 8; ++r) {
                    float v = acc[r] + bias;
                    v = v > 0.f ? v : 0.f;
                    bufA[(m0 + r) * 64 + n] = f2bf(v);
                }
            }
        }
        asm volatile("s_wait_dscnt 0x0" ::: "memory");

        // ---- layer 3: [32x64] @ [64x128], masked max over neighbors -> outX ----
        for (int nt = 0; nt < 8; ++nt) {
            float part = -FINF;
            for (int mt = 0; mt < 2; ++mt) {
                v8f acc = {};
#pragma unroll
                for (int ks = 0; ks < 2; ++ks) {
                    v16bf af = ldsA(bufA, 64, mt, ks, lane);
                    v16bf bf = ldsB(sW3, 64, nt, ks, lane);
                    acc = __builtin_amdgcn_wmma_f32_16x16x32_bf16(false, af, false, bf,
                                                                  (short)0, acc, false, false);
                }
                int m0 = mt * 16 + ((lane >= 16) ? 8 : 0);
#pragma unroll
                for (int r = 0; r < 8; ++r) {
                    int j = m0 + r;                                // neighbor index = row
                    float h = ((vmask >> j) & 1u) ? acc[r] : -FINF;
                    part = fmaxf(part, h);
                }
            }
            part = fmaxf(part, __shfl_xor(part, 16));              // combine lane L / L+16
            if (lane < 16) {
                int n = nt * 16 + lane;
                outX[(size_t)c * 128 + n] = part + b3[n];
            }
        }
        if (c + 2048 < NC) __builtin_prefetch(&nidx[(c + 2048) * KNB + lane], 0, 1);
    }
}

extern "C" void kernel_launch(void* const* d_in, const int* in_sizes, int n_in,
                              void* d_out, int out_size, void* d_ws, size_t ws_size,
                              hipStream_t stream) {
    const float* x   = (const float*)d_in[0];
    const float* pos = (const float*)d_in[1];
    // d_in[2] = batch (implicit: equal-size sorted clouds), d_in[3] = num_samples (=32)
    const float* W1 = (const float*)d_in[4];
    const float* b1 = (const float*)d_in[5];
    const float* W2 = (const float*)d_in[6];
    const float* b2 = (const float*)d_in[7];
    const float* W3 = (const float*)d_in[8];
    const float* b3 = (const float*)d_in[9];

    float* outX     = (float*)d_out;                 // [16384,128]
    float* outPos   = outX + (size_t)NC * 128;       // [16384,3]
    float* outBatch = outPos + (size_t)NC * 3;       // [16384]
    int*   nidx     = (int*)d_ws;                    // [16384,32] neighbor ids (-1 = invalid)

    fps_kernel<<<NB, 1024, 0, stream>>>(pos, outPos, outBatch);
    radius_kernel<<<(NC * 32) / 256, 256, 0, stream>>>(pos, outPos, nidx);
    pointconv_kernel<<<256, 256, 0, stream>>>(x, pos, outPos, nidx,
                                              W1, b1, W2, b2, W3, b3, outX);
}
